// AgnNetwork_24670292149145
// MI455X (gfx1250) — matrix operand order repaired and checked by reading, not compiled, not run on hardware
//
#include <hip/hip_runtime.h>

typedef __attribute__((ext_vector_type(16))) __bf16 v16bf;
typedef __attribute__((ext_vector_type(8)))  float  v8f;

#define N_NODES 100000
#define E_EDGES 1600000
#define EP      (E_EDGES + N_NODES)   // edges + self loops
#define F_IN    128
#define H_DIM   64
#define C_OUT   10
#define N_TILES (N_NODES / 16)        // 6250 exact

// ---------------------------------------------------------------------------
// GEMM1: h[N,64] = relu(x[N,128] @ W1[128,64] + b1), bf16 WMMA, fp32 accum.
// One wave per 16-row tile; 4 column tiles of 16; K = 4 steps of 32.
// ---------------------------------------------------------------------------
__global__ __launch_bounds__(128) void k_gemm1_relu(
    const float* __restrict__ x, const float* __restrict__ W1,
    const float* __restrict__ b1, float* __restrict__ h)
{
    int wave = blockIdx.x * (blockDim.x >> 5) + (threadIdx.x >> 5);
    if (wave >= N_TILES) return;
    int lane = threadIdx.x & 31;
    int m    = lane & 15;
    int half = lane >> 4;
    int row0 = wave * 16;

    v8f acc[4];
    #pragma unroll
    for (int nt = 0; nt < 4; ++nt) {
        float bias = b1[nt * 16 + m];
        #pragma unroll
        for (int v = 0; v < 8; ++v) acc[nt][v] = bias;
    }

    const float* arow = x + (size_t)(row0 + m) * F_IN;
    for (int kt = 0; kt < 4; ++kt) {
        int k0 = kt * 32;
        // A fragment: 16x32 bf16, ISA layout (VGPR j<4: K=half*8+2j; j>=4: +16)
        v16bf a;
        #pragma unroll
        for (int j = 0; j < 4; ++j) {
            float2 f = *(const float2*)(arow + k0 + half * 8 + 2 * j);
            a[2 * j]     = (__bf16)f.x;
            a[2 * j + 1] = (__bf16)f.y;
            float2 g = *(const float2*)(arow + k0 + 16 + half * 8 + 2 * j);
            a[8 + 2 * j]     = (__bf16)g.x;
            a[8 + 2 * j + 1] = (__bf16)g.y;
        }
        #pragma unroll
        for (int nt = 0; nt < 4; ++nt) {
            int col = nt * 16 + m;
            // B fragment: 32x16 bf16, same K layout as A (j<4: K=half*8+2j; j>=4: +16)
            v16bf b;
            #pragma unroll
            for (int j = 0; j < 8; ++j) {
                int k = k0 + (j >> 2) * 16 + half * 8 + 2 * (j & 3);
                b[2 * j]     = (__bf16)W1[(size_t)k * H_DIM + col];
                b[2 * j + 1] = (__bf16)W1[(size_t)(k + 1) * H_DIM + col];
            }
            acc[nt] = __builtin_amdgcn_wmma_f32_16x16x32_bf16(
                false, a, false, b, (short)0, acc[nt], false, false);
        }
    }
    // C layout: row = v + 8*half, col = lane&15 (+ nt*16)
    #pragma unroll
    for (int nt = 0; nt < 4; ++nt) {
        int col = nt * 16 + m;
        #pragma unroll
        for (int v = 0; v < 8; ++v) {
            int row = row0 + v + 8 * half;
            float z = acc[nt][v];
            h[(size_t)row * H_DIM + col] = z > 0.f ? z : 0.f;
        }
    }
}

// ---------------------------------------------------------------------------
// Per-layer init: zero aggregation target, reset segment max / denom.
// ---------------------------------------------------------------------------
__global__ void k_init(float* __restrict__ hout, float* __restrict__ amax,
                       float* __restrict__ denom)
{
    int i = blockIdx.x * blockDim.x + threadIdx.x;
    if (i < N_NODES * H_DIM) hout[i] = 0.f;
    if (i < N_NODES) { amax[i] = -2.0f; denom[i] = 0.f; }  // alpha in [-1,1]
}

// ---------------------------------------------------------------------------
// Row L2-normalize: one wave per node, lane = 2 channels (float2).
// ---------------------------------------------------------------------------
__global__ void k_normalize(const float* __restrict__ h, float* __restrict__ xn)
{
    int node = blockIdx.x * (blockDim.x >> 5) + (threadIdx.x >> 5);
    if (node >= N_NODES) return;
    int lane = threadIdx.x & 31;
    float2 v = *(const float2*)(h + (size_t)node * H_DIM + lane * 2);
    float s = v.x * v.x + v.y * v.y;
    #pragma unroll
    for (int off = 16; off >= 1; off >>= 1) s += __shfl_xor(s, off, 32);
    float r = rsqrtf(fmaxf(s, 1e-24f));
    float2 o; o.x = v.x * r; o.y = v.y * r;
    *(float2*)(xn + (size_t)node * H_DIM + lane * 2) = o;
}

// ---------------------------------------------------------------------------
// Pass A: alpha[e] = <xn[src], xn[dst]>; segment max via float atomicMax.
// One wave per edge.
// ---------------------------------------------------------------------------
__global__ void k_alpha(const float* __restrict__ xn,
                        const long long* __restrict__ srcE,
                        const long long* __restrict__ dstE,
                        float* __restrict__ alpha, float* __restrict__ amax)
{
    int e = blockIdx.x * (blockDim.x >> 5) + (threadIdx.x >> 5);
    if (e >= EP) return;
    int lane = threadIdx.x & 31;
    int s, d;
    if (e < E_EDGES) { s = (int)srcE[e]; d = (int)dstE[e]; }
    else             { s = d = e - E_EDGES; }   // self loop
    float2 a = *(const float2*)(xn + (size_t)s * H_DIM + lane * 2);
    float2 b = *(const float2*)(xn + (size_t)d * H_DIM + lane * 2);
    float p = a.x * b.x + a.y * b.y;
    #pragma unroll
    for (int off = 16; off >= 1; off >>= 1) p += __shfl_xor(p, off, 32);
    if (lane == 0) {
        alpha[e] = p;
        atomicMax(amax + d, p);
    }
}

// ---------------------------------------------------------------------------
// Pass B: ew = exp(alpha - amax[dst]); denom[dst] += ew. Thread per edge.
// ---------------------------------------------------------------------------
__global__ void k_expsum(const long long* __restrict__ dstE,
                         float* __restrict__ alpha_ew,
                         const float* __restrict__ amax,
                         float* __restrict__ denom)
{
    int e = blockIdx.x * blockDim.x + threadIdx.x;
    if (e >= EP) return;
    int d = (e < E_EDGES) ? (int)dstE[e] : (e - E_EDGES);
    float ew = __expf(alpha_ew[e] - amax[d]);
    alpha_ew[e] = ew;
    atomicAdd(denom + d, ew);
}

// ---------------------------------------------------------------------------
// Pass C: hout[dst] += (ew/denom[dst]) * h[src]. Wave per edge, lane = 2 ch.
// ---------------------------------------------------------------------------
__global__ void k_scatter(const float* __restrict__ h,
                          const long long* __restrict__ srcE,
                          const long long* __restrict__ dstE,
                          const float* __restrict__ ew,
                          const float* __restrict__ denom,
                          float* __restrict__ hout)
{
    int e = blockIdx.x * (blockDim.x >> 5) + (threadIdx.x >> 5);
    if (e >= EP) return;
    int lane = threadIdx.x & 31;
    int s, d;
    if (e < E_EDGES) { s = (int)srcE[e]; d = (int)dstE[e]; }
    else             { s = d = e - E_EDGES; }
    float w = ew[e] / fmaxf(denom[d], 1e-16f);
    float2 v = *(const float2*)(h + (size_t)s * H_DIM + lane * 2);
    atomicAdd(hout + (size_t)d * H_DIM + lane * 2,     w * v.x);
    atomicAdd(hout + (size_t)d * H_DIM + lane * 2 + 1, w * v.y);
}

// ---------------------------------------------------------------------------
// GEMM2 + log_softmax: logits[N,10] = h[N,64]@W2[64,10]+b2, pad N-dim to 16.
// One WMMA tile per wave (K = 2 steps of 32). Softmax reduces over 16-lane
// half-waves (row r's 16 cols live in one VGPR across 16 lanes).
// ---------------------------------------------------------------------------
__global__ __launch_bounds__(128) void k_gemm2_logsoftmax(
    const float* __restrict__ h, const float* __restrict__ W2,
    const float* __restrict__ b2, float* __restrict__ out)
{
    int wave = blockIdx.x * (blockDim.x >> 5) + (threadIdx.x >> 5);
    if (wave >= N_TILES) return;
    int lane = threadIdx.x & 31;
    int m    = lane & 15;
    int half = lane >> 4;
    int row0 = wave * 16;
    bool valid = (m < C_OUT);

    v8f acc;
    float bias = valid ? b2[m] : 0.f;
    #pragma unroll
    for (int v = 0; v < 8; ++v) acc[v] = bias;

    const float* arow = h + (size_t)(row0 + m) * H_DIM;
    #pragma unroll
    for (int kt = 0; kt < 2; ++kt) {
        int k0 = kt * 32;
        v16bf a, b;
        #pragma unroll
        for (int j = 0; j < 4; ++j) {
            float2 f = *(const float2*)(arow + k0 + half * 8 + 2 * j);
            a[2 * j]     = (__bf16)f.x;
            a[2 * j + 1] = (__bf16)f.y;
            float2 g = *(const float2*)(arow + k0 + 16 + half * 8 + 2 * j);
            a[8 + 2 * j]     = (__bf16)g.x;
            a[8 + 2 * j + 1] = (__bf16)g.y;
        }
        #pragma unroll
        for (int j = 0; j < 8; ++j) {
            int k = k0 + (j >> 2) * 16 + half * 8 + 2 * (j & 3);
            float w0 = valid ? W2[(size_t)k * C_OUT + m]       : 0.f;
            float w1 = valid ? W2[(size_t)(k + 1) * C_OUT + m] : 0.f;
            b[2 * j]     = (__bf16)w0;
            b[2 * j + 1] = (__bf16)w1;
        }
        acc = __builtin_amdgcn_wmma_f32_16x16x32_bf16(
            false, a, false, b, (short)0, acc, false, false);
    }

    #pragma unroll
    for (int v = 0; v < 8; ++v) {
        float z  = acc[v];
        float mx = valid ? z : -3.402823466e38f;
        #pragma unroll
        for (int off = 8; off >= 1; off >>= 1) mx = fmaxf(mx, __shfl_xor(mx, off, 16));
        float ex = valid ? __expf(z - mx) : 0.f;
        float sm = ex;
        #pragma unroll
        for (int off = 8; off >= 1; off >>= 1) sm += __shfl_xor(sm, off, 16);
        float o = (z - mx) - __logf(sm);
        if (valid) {
            int row = row0 + v + 8 * half;
            out[(size_t)row * C_OUT + m] = o;
        }
    }
}

// ---------------------------------------------------------------------------
extern "C" void kernel_launch(void* const* d_in, const int* in_sizes, int n_in,
                              void* d_out, int out_size, void* d_ws, size_t ws_size,
                              hipStream_t stream)
{
    const float*     x   = (const float*)d_in[0];
    const long long* ei  = (const long long*)d_in[1];   // int64 [2, E] flat
    const float*     W1  = (const float*)d_in[2];
    const float*     b1  = (const float*)d_in[3];
    const float*     W2  = (const float*)d_in[4];
    const float*     b2  = (const float*)d_in[5];
    const long long* srcE = ei;
    const long long* dstE = ei + E_EDGES;

    char*  ws  = (char*)d_ws;
    size_t off = 0;
    auto alloc = [&](size_t bytes) -> void* {
        void* p = ws + off;
        off += (bytes + 255) & ~(size_t)255;
        return p;
    };
    float* hA    = (float*)alloc((size_t)N_NODES * H_DIM * 4);
    float* hB    = (float*)alloc((size_t)N_NODES * H_DIM * 4);
    float* xn    = (float*)alloc((size_t)N_NODES * H_DIM * 4);
    float* alpha = (float*)alloc((size_t)EP * 4);
    float* amax  = (float*)alloc((size_t)N_NODES * 4);
    float* denom = (float*)alloc((size_t)N_NODES * 4);

    // h = relu(x @ W1 + b1)
    k_gemm1_relu<<<(N_TILES + 3) / 4, 128, 0, stream>>>(x, W1, b1, hA);

    for (int l = 0; l < 4; ++l) {
        k_init<<<(N_NODES * H_DIM + 255) / 256, 256, 0, stream>>>(hB, amax, denom);
        k_normalize<<<(N_NODES + 7) / 8, 256, 0, stream>>>(hA, xn);
        k_alpha<<<(EP + 7) / 8, 256, 0, stream>>>(xn, srcE, dstE, alpha, amax);
        k_expsum<<<(EP + 255) / 256, 256, 0, stream>>>(dstE, alpha, amax, denom);
        k_scatter<<<(EP + 7) / 8, 256, 0, stream>>>(hA, srcE, dstE, alpha, denom, hB);
        float* t = hA; hA = hB; hB = t;
    }

    // logits + log_softmax
    k_gemm2_logsoftmax<<<(N_TILES + 3) / 4, 128, 0, stream>>>(hA, W2, b2, (float*)d_out);
}
